// LegendreFeatureLayer_2654289789589
// MI455X (gfx1250) — matrix operand order repaired and checked by reading, not compiled, not run on hardware
//
#include <hip/hip_runtime.h>
#include <hip/hip_bf16.h>
#include <stdint.h>

// ---------------------------------------------------------------------------
// LMU scan as 64 dependent GEMMs of shape (64 x 16384 x 544) using
// v_wmma_f32_16x16x32_bf16 (gfx1250, wave32), with log-depth f32 precompute
// of A^k and fused state/u-tap packing (double-buffered Z).
// ---------------------------------------------------------------------------

#define DD    512            // state dim
#define KCH   32             // time steps per chunk
#define CPAD  544            // 512 state + 32 u-taps (= 17 * 32, no waste)
#define NB    64             // batch
#define TT    2048           // sequence length
#define NCHUNK (TT / KCH)    // 64

typedef __attribute__((ext_vector_type(16))) __bf16 v16bf;
typedef __attribute__((ext_vector_type(8)))  float  v8f;

union Frag { uint4 u[2]; v16bf v; };

// round-to-nearest-even f32 -> bf16 via bit ops
__device__ __forceinline__ unsigned short f2bf(float f) {
    union { float f; unsigned u; } v; v.f = f;
    unsigned r = v.u + 0x7FFFu + ((v.u >> 16) & 1u);
    return (unsigned short)(r >> 16);
}

__device__ __forceinline__ v8f wmma_bf16(const Frag& a, const Frag& b, v8f c) {
    return __builtin_amdgcn_wmma_f32_16x16x32_bf16(
        false, a.v, false, b.v, (short)0, c, false, false);
}

// ---------------------------------------------------------------------------
// Precompute (f32 — errors here would be baked into every chunk)
// ---------------------------------------------------------------------------

// P_1 = A (f32 copy) ; W[0] state block = bf16(A)
__global__ void lmu_convA(const float* __restrict__ A, float* __restrict__ Pows,
                          unsigned short* __restrict__ Wst) {
    int idx = blockIdx.x * blockDim.x + threadIdx.x;   // 512*512
    int n = idx >> 9, c = idx & (DD - 1);
    float v = A[idx];
    Pows[idx] = v;
    Wst[(size_t)n * CPAD + c] = f2bf(v);
}

// Batched doubling step: for j = 1..m (blockIdx.z), P_{m+j} = P_j @ P_m  (f32),
// also rounding into Wstack[m+j-1] state block. 64x64 tile, 4x4 per thread.
__global__ void lmu_powgemm(const float* __restrict__ Pows, int m,
                            float* __restrict__ PowsOut,
                            unsigned short* __restrict__ Wst) {
    const int j = blockIdx.z + 1;
    const float* Pj = Pows + (size_t)(j - 1) * DD * DD;
    const float* Pm = Pows + (size_t)(m - 1) * DD * DD;
    float* C = PowsOut + (size_t)(m + j - 1) * DD * DD;
    unsigned short* Wk = Wst + (size_t)(m + j - 1) * DD * CPAD;

    __shared__ float sa[16][65];   // [kk][row]
    __shared__ float sb[16][65];   // [kk][col]
    const int tid = threadIdx.x;
    const int tx = tid & 15, ty = tid >> 4;
    const int R0 = blockIdx.y * 64, C0 = blockIdx.x * 64;

    float acc[4][4] = {};
    for (int kt = 0; kt < DD; kt += 16) {
        {   // A tile: 64 rows x 16 k
            int rr = tid >> 2, kk0 = (tid & 3) * 4;
            float4 av = *(const float4*)(Pj + (size_t)(R0 + rr) * DD + kt + kk0);
            sa[kk0 + 0][rr] = av.x; sa[kk0 + 1][rr] = av.y;
            sa[kk0 + 2][rr] = av.z; sa[kk0 + 3][rr] = av.w;
            // B tile: 16 k x 64 cols
            int kk = tid >> 4, cc0 = (tid & 15) * 4;
            float4 bv = *(const float4*)(Pm + (size_t)(kt + kk) * DD + C0 + cc0);
            sb[kk][cc0 + 0] = bv.x; sb[kk][cc0 + 1] = bv.y;
            sb[kk][cc0 + 2] = bv.z; sb[kk][cc0 + 3] = bv.w;
        }
        __syncthreads();
#pragma unroll
        for (int q = 0; q < 16; ++q) {
            float ar[4], br[4];
#pragma unroll
            for (int i = 0; i < 4; ++i) ar[i] = sa[q][ty * 4 + i];
#pragma unroll
            for (int i = 0; i < 4; ++i) br[i] = sb[q][tx * 4 + i];
#pragma unroll
            for (int i = 0; i < 4; ++i)
#pragma unroll
                for (int jj = 0; jj < 4; ++jj) acc[i][jj] += ar[i] * br[jj];
        }
        __syncthreads();
    }
#pragma unroll
    for (int i = 0; i < 4; ++i)
#pragma unroll
        for (int jj = 0; jj < 4; ++jj) {
            int row = R0 + ty * 4 + i, col = C0 + tx * 4 + jj;
            C[(size_t)row * DD + col] = acc[i][jj];
            Wk[(size_t)row * CPAD + col] = f2bf(acc[i][jj]);
        }
}

// h_m = A^m @ Bd for m = 0..31 (h_0 = Bd), all m in parallel
__global__ void lmu_hvec(const float* __restrict__ Pows, const float* __restrict__ Bd,
                         float* __restrict__ h) {
    int idx = blockIdx.x * blockDim.x + threadIdx.x;   // 32*512
    int m = idx >> 9, i = idx & (DD - 1);
    float s;
    if (m == 0) {
        s = Bd[i];
    } else {
        const float* P = Pows + (size_t)(m - 1) * DD * DD + (size_t)i * DD;
        s = 0.f;
        for (int jj = 0; jj < DD; ++jj) s += P[jj] * Bd[jj];
    }
    h[idx] = s;
}

// W[kIdx][n][512+jj] = (jj <= kIdx) ? h_{kIdx-jj}[n] : 0
__global__ void lmu_fillG(unsigned short* __restrict__ W, const float* __restrict__ h) {
    int idx = blockIdx.x * blockDim.x + threadIdx.x;   // 32*512*32
    int kIdx = idx >> 14;
    int n    = (idx >> 5) & (DD - 1);
    int jj   = idx & 31;
    float v = (jj <= kIdx) ? h[(size_t)(kIdx - jj) * DD + n] : 0.f;
    W[((size_t)kIdx * DD + n) * CPAD + DD + jj] = f2bf(v);
}

// Z for chunk 0: [ 0 state | u_{1..32} ]
__global__ void lmu_packZ0(unsigned short* __restrict__ Z, const float* __restrict__ xin) {
    int idx = blockIdx.x * blockDim.x + threadIdx.x;   // 64*544
    int b = idx / CPAD, c = idx % CPAD;
    float v = (c < DD) ? 0.f : xin[(size_t)b * TT + (c - DD)];
    Z[idx] = f2bf(v);
}

// ---------------------------------------------------------------------------
// Chunk kernel: Y(64 x 16384) = Z(64x544) @ W^T. One wave per 16-wide n-tile,
// register-blocked over all 4 batch tiles (68 WMMAs/wave, B reused 4x).
// Z staged in LDS (68 KB); next chunk's Z written directly (double buffer).
// ---------------------------------------------------------------------------
__global__ void lmu_chunk(const unsigned short* __restrict__ Zr,
                          unsigned short* __restrict__ Zw,
                          const unsigned short* __restrict__ W,
                          const float* __restrict__ xin,
                          float* __restrict__ out,
                          int t0, int last) {
    extern __shared__ unsigned short zs[];             // 64 * 544 bf16 = 68 KB

    // cooperative copy of Z into LDS (4352 x 16B)
    for (int i = threadIdx.x; i < NB * CPAD / 8; i += blockDim.x)
        ((uint4*)zs)[i] = ((const uint4*)Zr)[i];
    __syncthreads();

    const int wave   = (blockIdx.x * blockDim.x + threadIdx.x) >> 5;  // 0..1023
    const int lane   = threadIdx.x & 31;
    const int laneLo = lane & 15;
    const int laneHi = lane >> 4;
    const int kIdx   = wave >> 5;            // 0..31 : time step in chunk
    const int nBase  = (wave & 31) * 16;     // state-dim tile

    const unsigned short* wrow = W + ((size_t)kIdx * DD + nBase + laneLo) * CPAD;
    const unsigned short* za   = zs + (size_t)laneLo * CPAD;
    __builtin_prefetch(wrow, 0, 1);          // global_prefetch_b8

    const int aOff = laneHi * 8;             // A-frag: runs at +aOff, +aOff+16
    const int bOff = laneHi * 8;             // B-frag: runs at +bOff, +bOff+16

    v8f acc0 = {}, acc1 = {}, acc2 = {}, acc3 = {};
#pragma unroll
    for (int kk = 0; kk < CPAD / 32; ++kk) { // 17 k-steps
        Frag b;
        const int cB = kk * 32 + bOff;
        b.u[0] = *(const uint4*)(wrow + cB);
        b.u[1] = *(const uint4*)(wrow + cB + 16);
        const int cA = kk * 32 + aOff;
        Frag a0, a1, a2, a3;
        a0.u[0] = *(const uint4*)(za + cA);
        a0.u[1] = *(const uint4*)(za + cA + 16);
        a1.u[0] = *(const uint4*)(za + 16 * CPAD + cA);
        a1.u[1] = *(const uint4*)(za + 16 * CPAD + cA + 16);
        a2.u[0] = *(const uint4*)(za + 32 * CPAD + cA);
        a2.u[1] = *(const uint4*)(za + 32 * CPAD + cA + 16);
        a3.u[0] = *(const uint4*)(za + 48 * CPAD + cA);
        a3.u[1] = *(const uint4*)(za + 48 * CPAD + cA + 16);
        acc0 = wmma_bf16(a0, b, acc0);       // 4 independent chains hide
        acc1 = wmma_bf16(a1, b, acc1);       // XDL latency / RAW NOPs
        acc2 = wmma_bf16(a2, b, acc2);
        acc3 = wmma_bf16(a3, b, acc3);
    }

    // D layout: lane%16 = N, VGPR r -> M = r + laneHi*8
    const int n = nBase + laneLo;
    const int t = t0 + kIdx;
    const int mH = laneHi * 8;
    const bool isLastStep = (kIdx == KCH - 1);
#pragma unroll
    for (int r = 0; r < 8; ++r) {
        const int m0 = mH + r;
        out[((size_t)(m0     ) * TT + t) * DD + n] = acc0[r];
        out[((size_t)(m0 + 16) * TT + t) * DD + n] = acc1[r];
        out[((size_t)(m0 + 32) * TT + t) * DD + n] = acc2[r];
        out[((size_t)(m0 + 48) * TT + t) * DD + n] = acc3[r];
        if (isLastStep) {                    // becomes next chunk's state
            Zw[(size_t)(m0     ) * CPAD + n] = f2bf(acc0[r]);
            Zw[(size_t)(m0 + 16) * CPAD + n] = f2bf(acc1[r]);
            Zw[(size_t)(m0 + 32) * CPAD + n] = f2bf(acc2[r]);
            Zw[(size_t)(m0 + 48) * CPAD + n] = f2bf(acc3[r]);
        }
    }

    // one wave packs next chunk's u-taps: Zw[b][512+jj] = x[b][t0+32+jj]
    if (wave == 0 && !last) {
        for (int b = 0; b < NB; ++b)
            Zw[(size_t)b * CPAD + DD + lane] = f2bf(xin[(size_t)b * TT + t0 + KCH + lane]);
    }
}

// ---------------------------------------------------------------------------
// Host launcher
// ---------------------------------------------------------------------------
extern "C" void kernel_launch(void* const* d_in, const int* in_sizes, int n_in,
                              void* d_out, int out_size, void* d_ws, size_t ws_size,
                              hipStream_t stream) {
    (void)in_sizes; (void)n_in; (void)out_size; (void)ws_size;

    const float* xin = (const float*)d_in[0];   // (64, 2048, 1) f32
    const float* A   = (const float*)d_in[1];   // (512, 512)  f32 (discretized Ad)
    const float* Bd  = (const float*)d_in[2];   // (512, 1)    f32 (Bd)
    float*       out = (float*)d_out;           // (64, 2048, 512) f32

    // workspace layout
    char* ws = (char*)d_ws;
    constexpr size_t W_BYTES = (size_t)KCH * DD * CPAD * sizeof(unsigned short); // 17.8 MB
    constexpr size_t P_BYTES = (size_t)KCH * DD * DD * sizeof(float);            // 32 MB
    constexpr size_t H_BYTES = (size_t)KCH * DD * sizeof(float);                 // 64 KB
    constexpr size_t Z_BYTES = (size_t)NB * CPAD * sizeof(unsigned short);       // 68 KB

    unsigned short* Wst  = (unsigned short*)(ws);
    float*          Pows = (float*)(ws + W_BYTES);
    float*          hbuf = (float*)(ws + W_BYTES + P_BYTES);
    unsigned short* Z0   = (unsigned short*)(ws + W_BYTES + P_BYTES + H_BYTES);
    unsigned short* Z1   = (unsigned short*)(ws + W_BYTES + P_BYTES + H_BYTES + Z_BYTES);

    // ---- precompute: A^k via log-depth doubling (5 batched launches) ----
    lmu_convA<<<(DD * DD) / 256, 256, 0, stream>>>(A, Pows, Wst);
    for (int m = 1; m < KCH; m <<= 1)       // m = 1,2,4,8,16 -> powers 2..32
        lmu_powgemm<<<dim3(DD / 64, DD / 64, m), 256, 0, stream>>>(Pows, m, Pows, Wst);
    lmu_hvec<<<(KCH * DD) / 256, 256, 0, stream>>>(Pows, Bd, hbuf);
    lmu_fillG<<<(KCH * DD * 32) / 256, 256, 0, stream>>>(Wst, hbuf);
    lmu_packZ0<<<(NB * CPAD + 255) / 256, 256, 0, stream>>>(Z0, xin);

    // ---- 64 dependent chunk GEMMs (the scan), double-buffered Z ----
    const size_t ldsBytes = (size_t)NB * CPAD * sizeof(unsigned short);
    for (int chunk = 0; chunk < NCHUNK; ++chunk) {
        unsigned short* Zr = (chunk & 1) ? Z1 : Z0;
        unsigned short* Zw = (chunk & 1) ? Z0 : Z1;
        lmu_chunk<<<128, 256, ldsBytes, stream>>>(
            Zr, Zw, Wst, xin, out, chunk * KCH, chunk == NCHUNK - 1 ? 1 : 0);
    }
}
